// Linear8bitLt_1726576855766
// MI455X (gfx1250) — 8-bit matrix operand order repaired and checked by reading, compiled, not run on hardware
//
#include <hip/hip_runtime.h>
#include <hip/hip_bf16.h>
#include <stdint.h>

typedef int v8i __attribute__((ext_vector_type(8)));

#define THRESH 6.0f
#define TTOK   8192
#define KDIM   2048
#define NDIM   2048

__global__ void wquant_kernel(const float* __restrict__ w,
                              int8_t* __restrict__ wq,
                              float* __restrict__ wscale) {
  __shared__ float red[256];
  const int row = blockIdx.x;
  const int tid = threadIdx.x;
  const float* wr = w + (size_t)row * KDIM;
  float vals[KDIM / 256];
  float m = 0.0f;
#pragma unroll
  for (int i = 0; i < KDIM / 256; ++i) {
    float v = wr[tid + i * 256];
    vals[i] = v;
    m = fmaxf(m, fabsf(v));
  }
  red[tid] = m;
  __syncthreads();
  for (int s = 128; s > 0; s >>= 1) {
    if (tid < s) red[tid] = fmaxf(red[tid], red[tid + s]);
    __syncthreads();
  }
  const float absmax = fmaxf(red[0], 1e-6f);
  const float inv = 127.0f / absmax;
  int8_t* qr = wq + (size_t)row * KDIM;
#pragma unroll
  for (int i = 0; i < KDIM / 256; ++i) {
    float q = rintf(vals[i] * inv);
    q = fminf(fmaxf(q, -127.0f), 127.0f);
    qr[tid + i * 256] = (int8_t)q;
  }
  if (tid == 0) wscale[row] = absmax * (1.0f / 127.0f);
}

__global__ void colmax_init_kernel(unsigned* __restrict__ colbits,
                                   int* __restrict__ nout) {
  const int gid = blockIdx.x * 256 + threadIdx.x;
  if (gid < KDIM) colbits[gid] = 0u;
  if (gid == 0) nout[0] = 0;
}

__global__ void colmax_kernel(const float* __restrict__ x,
                              unsigned* __restrict__ colbits) {
  const int r0 = blockIdx.x * 128;
  for (int c = threadIdx.x; c < KDIM; c += 256) {
    float m = 0.0f;
    for (int r = 0; r < 128; ++r)
      m = fmaxf(m, fabsf(x[(size_t)(r0 + r) * KDIM + c]));
    atomicMax(colbits + c, __float_as_uint(m));
  }
}

__global__ void outlier_kernel(const unsigned* __restrict__ colbits,
                               unsigned char* __restrict__ omask,
                               int* __restrict__ oidx,
                               int* __restrict__ nout) {
  const int j = blockIdx.x * 256 + threadIdx.x;
  if (j >= KDIM) return;
  const float cm = __uint_as_float(colbits[j]);
  const bool o = cm > THRESH;
  omask[j] = o ? 1 : 0;
  if (o) {
    int p = atomicAdd(nout, 1);
    oidx[p] = j;
  }
}

__global__ void xquant_kernel(const float* __restrict__ x,
                              const unsigned char* __restrict__ omask,
                              int8_t* __restrict__ xq,
                              float* __restrict__ xscale) {
  __shared__ float red[256];
  const int t = blockIdx.x;
  const int tid = threadIdx.x;
  const float* xr = x + (size_t)t * KDIM;
  float vals[KDIM / 256];
  float m = 0.0f;
#pragma unroll
  for (int i = 0; i < KDIM / 256; ++i) {
    const int c = tid + i * 256;
    float v = omask[c] ? 0.0f : xr[c];
    vals[i] = v;
    m = fmaxf(m, fabsf(v));
  }
  red[tid] = m;
  __syncthreads();
  for (int s = 128; s > 0; s >>= 1) {
    if (tid < s) red[tid] = fmaxf(red[tid], red[tid + s]);
    __syncthreads();
  }
  const float absmax = fmaxf(red[0], 1e-6f);
  const float inv = 127.0f / absmax;
  int8_t* qr = xq + (size_t)t * KDIM;
#pragma unroll
  for (int i = 0; i < KDIM / 256; ++i) {
    float q = rintf(vals[i] * inv);
    q = fminf(fmaxf(q, -127.0f), 127.0f);
    qr[tid + i * 256] = (int8_t)q;
  }
  if (tid == 0) xscale[t] = absmax * (1.0f / 127.0f);
}

__device__ __forceinline__ v8i load_a_frag(const int* p  ) {
  int2 c0 = *(const int2*)(p);
  int2 c1 = *(const int2*)(p + 2);
  int2 c2 = *(const int2*)(p + 8);
  int2 c3 = *(const int2*)(p + 10);
  v8i a;
  a[0] = c0.x; a[1] = c0.y; a[2] = c1.x; a[3] = c1.y;
  a[4] = c2.x; a[5] = c2.y; a[6] = c3.x; a[7] = c3.y;
  return a;
}
__device__ __forceinline__ v8i load_b_frag(const int* p  ) {
  int4 d0 = *(const int4*)(p);
  int4 d1 = *(const int4*)(p + 8);
  v8i b;
  b[0] = d0.x; b[1] = d0.y; b[2] = d0.z; b[3] = d0.w;
  b[4] = d1.x; b[5] = d1.y; b[6] = d1.z; b[7] = d1.w;
  return b;
}

__device__ __forceinline__ void epilogue_tile(
    v8i acc, int mbase, int n, const float* __restrict__ xscale, float wsc,
    float bv, const float* __restrict__ x, const int8_t* __restrict__ wqn,
    const int* __restrict__ oidx, int n_out, float* __restrict__ out) {
#pragma unroll
  for (int i = 0; i < 8; ++i) {
    const int m = mbase + i;
    float v = (float)acc[i] * xscale[m] * wsc + bv;
    const float* xrow = x + (size_t)m * KDIM;
    for (int j = 0; j < n_out; ++j) {
      const int c = oidx[j];
      v += xrow[c] * (float)wqn[c] * wsc;
    }
    out[(size_t)m * NDIM + n] = v;
  }
}

__global__ __launch_bounds__(256) void int8_gemm_kernel(
    const int8_t* __restrict__ xq, const int8_t* __restrict__ wq,
    const float* __restrict__ xscale, const float* __restrict__ wscale,
    const float* __restrict__ bias, const float* __restrict__ x,
    const int* __restrict__ oidx, const int* __restrict__ noutp,
    float* __restrict__ out) {
  const int lane = threadIdx.x & 31;
  const int wave = threadIdx.x >> 5;
  const int wm = wave >> 1;
  const int wn = wave & 1;
  const int Moff = blockIdx.y * 128 + wm * 32;
  const int Noff = blockIdx.x * 64 + wn * 32;
  const int lo = lane & 15;
  const int hi = lane >> 4;
  const int n_out = noutp[0];

  const int* ap0 = (const int*)(xq + (size_t)(Moff + lo) * KDIM);
  const int* ap1 = (const int*)(xq + (size_t)(Moff + 16 + lo) * KDIM);
  const int* bp0 = (const int*)(wq + (size_t)(Noff + lo) * KDIM);
  const int* bp1 = (const int*)(wq + (size_t)(Noff + 16 + lo) * KDIM);
  const int ao = hi * 4;
  const int bo = hi * 4;

  v8i acc00 = {}; v8i acc01 = {}; v8i acc10 = {}; v8i acc11 = {};

  for (int k0 = 0; k0 < KDIM; k0 += 64) {
    const int ko = k0 >> 2;
    v8i A0 = load_a_frag(ap0 + ko + ao);
    v8i A1 = load_a_frag(ap1 + ko + ao);
    v8i B0 = load_b_frag(bp0 + ko + bo);
    v8i B1 = load_b_frag(bp1 + ko + bo);
    acc00 = __builtin_amdgcn_wmma_i32_16x16x64_iu8(true, A0, true, B0, acc00, false, false);
    acc01 = __builtin_amdgcn_wmma_i32_16x16x64_iu8(true, A0, true, B1, acc01, false, false);
    acc10 = __builtin_amdgcn_wmma_i32_16x16x64_iu8(true, A1, true, B0, acc10, false, false);
    acc11 = __builtin_amdgcn_wmma_i32_16x16x64_iu8(true, A1, true, B1, acc11, false, false);
  }

  const int n0 = Noff + lo;
  const int n1 = Noff + 16 + lo;
  const float ws0 = wscale[n0], ws1 = wscale[n1];
  const float bv0 = bias[n0], bv1 = bias[n1];
  const int8_t* wqn0 = wq + (size_t)n0 * KDIM;
  const int8_t* wqn1 = wq + (size_t)n1 * KDIM;

  epilogue_tile(acc00, Moff + 8 * hi,      n0, xscale, ws0, bv0, x, wqn0, oidx, n_out, out);
  epilogue_tile(acc10, Moff + 16 + 8 * hi, n0, xscale, ws0, bv0, x, wqn0, oidx, n_out, out);
  epilogue_tile(acc01, Moff + 8 * hi,      n1, xscale, ws1, bv1, x, wqn1, oidx, n_out, out);
  epilogue_tile(acc11, Moff + 16 + 8 * hi, n1, xscale, ws1, bv1, x, wqn1, oidx, n_out, out);
}

extern "C" void kernel_launch(void* const* d_in, const int* in_sizes, int n_in,
                              void* d_out, int out_size, void* d_ws, size_t ws_size,
                              hipStream_t stream) {
  const float* x    = (const float*)d_in[0];
  const float* w    = (const float*)d_in[1];
  const float* bias = (const float*)d_in[2];
  float* out = (float*)d_out;

  char* ws = (char*)d_ws;
  size_t off = 0;
  auto carve = [&](size_t bytes) -> void* {
    off = (off + 255) & ~(size_t)255;
    void* p = ws + off;
    off += bytes;
    return p;
  };
  int8_t*  wq      = (int8_t*)carve((size_t)NDIM * KDIM);
  int8_t*  xq      = (int8_t*)carve((size_t)TTOK * KDIM);
  float*   wscale  = (float*)carve(NDIM * sizeof(float));
  float*   xscale  = (float*)carve(TTOK * sizeof(float));
  unsigned* colbits = (unsigned*)carve(KDIM * sizeof(unsigned));
  int*     oidx    = (int*)carve(KDIM * sizeof(int));
  int*     nout    = (int*)carve(256);
  unsigned char* omask = (unsigned char*)carve(KDIM);

  wquant_kernel<<<NDIM, 256, 0, stream>>>(w, wq, wscale);
  colmax_init_kernel<<<KDIM / 256, 256, 0, stream>>>(colbits, nout);
  colmax_kernel<<<TTOK / 128, 256, 0, stream>>>(x, colbits);
  outlier_kernel<<<KDIM / 256, 256, 0, stream>>>(colbits, omask, oidx, nout);
  xquant_kernel<<<TTOK, 256, 0, stream>>>(x, omask, xq, xscale);

  dim3 grid(NDIM / 64, TTOK / 128);
  int8_gemm_kernel<<<grid, 256, 0, stream>>>(xq, wq, xscale, wscale, bias, x,
                                             oidx, nout, out);
}
